// GINNet_68856915690108
// MI455X (gfx1250) — matrix operand order repaired and checked by reading, not compiled, not run on hardware
//
#include <hip/hip_runtime.h>
#include <math.h>

// ---------------------------------------------------------------------------
// GIN pipeline on gfx1250 (MI455X).
//  - fp32 V_WMMA_F32_16X16X4_F32 for all data-path GEMMs (keeps reference
//    precision on the X -> MLP -> BN chain).
//  - bf16 V_WMMA_F32_16X16X32_BF16 for S = h@h.T (8x matrix rate; S only
//    feeds the "> mean" threshold, so bf16 inputs are safe).
// ---------------------------------------------------------------------------

typedef float  v2f  __attribute__((ext_vector_type(2)));
typedef float  v8f  __attribute__((ext_vector_type(8)));
typedef __bf16 v16bf __attribute__((ext_vector_type(16)));

enum { AM_DIRECT = 0, AM_THRESH = 1, AM_BN = 2 };
enum { BM_DIRECT = 0, BM_TRANS = 1 };
enum { EP_PLAIN = 0, EP_BIAS = 1, EP_BIAS_ELU = 2, EP_GRAM = 3, EP_BIAS_ACC = 4 };

#define BLK_M 64
#define BLK_N 64
#define BLK_K 32

// ---------------------------------------------------------------------------
// Generic fp32 WMMA GEMM: C = epilogue( transform(A) @ transform(B) )
// Block: 256 threads = 8 waves. Wave w: rows (w%4)*16..+16, cols (w/4)*32..+32
// (two 16x16 accumulators). K staged through LDS in BK=32 slabs.
// ---------------------------------------------------------------------------
template <int AMODE, int BMODE, int EP>
__global__ __launch_bounds__(256) void wmma_gemm(
    const float* __restrict__ A, const float* __restrict__ B,
    float* __restrict__ C, const float* __restrict__ bias,
    const float* __restrict__ mu, const float* __restrict__ var,
    const float* __restrict__ gamma, const float* __restrict__ beta,
    const float* __restrict__ sumPtr, float meanScale,
    float* __restrict__ acc,
    int M, int Nn, int K, int lda, int ldb, int ldc) {
  __shared__ float As[BLK_M][BLK_K + 1];
  __shared__ float Bs[BLK_K][BLK_N + 1];

  const int tid  = threadIdx.x;
  const int lane = tid & 31;
  const int wave = tid >> 5;
  const int wm   = wave & 3;   // row sub-tile (16 rows)
  const int wn   = wave >> 2;  // col sub-tile (32 cols)
  const int bm   = blockIdx.y * BLK_M;
  const int bn   = blockIdx.x * BLK_N;

  float thr = 0.f;
  if (AMODE == AM_THRESH) thr = sumPtr[0] * meanScale;  // global mean of S

  v8f c0 = {};
  v8f c1 = {};

  for (int k0 = 0; k0 < K; k0 += BLK_K) {
    // ---- stage A tile [BLK_M x BLK_K] -----------------------------------
#pragma unroll
    for (int p = 0; p < 2; ++p) {
      const int r  = (tid >> 3) + p * 32;   // 0..63
      const int cc = (tid & 7) * 4;         // 0..28
      const int gr = bm + r;
      const int gk = k0 + cc;
      const float4 t = *(const float4*)(A + (size_t)gr * lda + gk);
      float v0 = t.x, v1 = t.y, v2 = t.z, v3 = t.w;
      if (AMODE == AM_THRESH) {
        // adj_onehot + E*I fused into the A operand
        v0 = (t.x > thr ? 1.f : 0.f) + ((gr == gk + 0) ? 1.f : 0.f);
        v1 = (t.y > thr ? 1.f : 0.f) + ((gr == gk + 1) ? 1.f : 0.f);
        v2 = (t.z > thr ? 1.f : 0.f) + ((gr == gk + 2) ? 1.f : 0.f);
        v3 = (t.w > thr ? 1.f : 0.f) + ((gr == gk + 3) ? 1.f : 0.f);
      } else if (AMODE == AM_BN) {
        // BatchNorm (training stats) + ReLU fused into the A operand
#pragma unroll
        for (int q = 0; q < 4; ++q) {
          const int j   = gk + q;
          const float x = (q == 0 ? t.x : q == 1 ? t.y : q == 2 ? t.z : t.w);
          float y = (x - mu[j]) * __frsqrt_rn(var[j] + 1e-5f) * gamma[j] + beta[j];
          y = y > 0.f ? y : 0.f;
          if (q == 0) v0 = y; else if (q == 1) v1 = y; else if (q == 2) v2 = y; else v3 = y;
        }
      }
      As[r][cc + 0] = v0; As[r][cc + 1] = v1; As[r][cc + 2] = v2; As[r][cc + 3] = v3;
    }
    // ---- stage B tile [BLK_K x BLK_N] -----------------------------------
    if (BMODE == BM_DIRECT) {
#pragma unroll
      for (int p = 0; p < 2; ++p) {
        const int kk = (tid >> 4) + p * 16;  // 0..31
        const int cc = (tid & 15) * 4;       // 0..60
        const float4 t = *(const float4*)(B + (size_t)(k0 + kk) * ldb + bn + cc);
        Bs[kk][cc + 0] = t.x; Bs[kk][cc + 1] = t.y;
        Bs[kk][cc + 2] = t.z; Bs[kk][cc + 3] = t.w;
      }
    } else {  // B_TRANS: Bs[k][c] = B[(bn+c)*ldb + k0+k]
#pragma unroll
      for (int p = 0; p < 2; ++p) {
        const int cc = (tid >> 3) + p * 32;  // 0..63
        const int kk = (tid & 7) * 4;        // 0..28
        const float4 t = *(const float4*)(B + (size_t)(bn + cc) * ldb + k0 + kk);
        Bs[kk + 0][cc] = t.x; Bs[kk + 1][cc] = t.y;
        Bs[kk + 2][cc] = t.z; Bs[kk + 3][cc] = t.w;
      }
    }
    // prefetch next K slab while we compute on this one
    if (k0 + BLK_K < K) {
      __builtin_prefetch(A + (size_t)(bm + (tid >> 3)) * lda + k0 + BLK_K, 0, 1);
      if (BMODE == BM_DIRECT)
        __builtin_prefetch(B + (size_t)(k0 + BLK_K + (tid >> 4)) * ldb + bn, 0, 1);
      else
        __builtin_prefetch(B + (size_t)(bn + (tid >> 3)) * ldb + k0 + BLK_K, 0, 1);
    }
    __syncthreads();

    // ---- WMMA inner loop: 8 steps of K=4, two 16x16 tiles per wave ------
    const int arow  = wm * 16 + (lane & 15);
    const int koff  = (lane >> 4) << 1;      // lanes 16-31 hold K+2,K+3
    const int bcol0 = wn * 32 + (lane & 15);
#pragma unroll
    for (int kk = 0; kk < BLK_K; kk += 4) {
      v2f a, b0, b1;
      a.x  = As[arow][kk + koff];
      a.y  = As[arow][kk + koff + 1];
      b0.x = Bs[kk + koff][bcol0];
      b0.y = Bs[kk + koff + 1][bcol0];
      b1.x = Bs[kk + koff][bcol0 + 16];
      b1.y = Bs[kk + koff + 1][bcol0 + 16];
      c0 = __builtin_amdgcn_wmma_f32_16x16x4_f32(false, a, false, b0, (short)0, c0, false, false);
      c1 = __builtin_amdgcn_wmma_f32_16x16x4_f32(false, a, false, b1, (short)0, c1, false, false);
    }
    __syncthreads();
  }

  // ---- epilogue ---------------------------------------------------------
  const int row0 = bm + wm * 16;
  const int col0 = bn + wn * 32;
#pragma unroll
  for (int r = 0; r < 8; ++r) {
    const int grow = row0 + r + ((lane >> 4) << 3);  // C layout: VGPR r -> M=r / M=r+8
    const int gc0  = col0 + (lane & 15);
    const int gc1  = gc0 + 16;
    float v0 = c0[r], v1 = c1[r];
    if (EP == EP_BIAS || EP == EP_BIAS_ELU || EP == EP_BIAS_ACC) {
      v0 += bias[gc0];
      v1 += bias[gc1];
    }
    if (EP == EP_BIAS_ELU) {  // ELU(alpha=1)
      v0 = v0 > 0.f ? v0 : __expf(v0) - 1.f;
      v1 = v1 > 0.f ? v1 : __expf(v1) - 1.f;
    }
    const size_t i0 = (size_t)grow * ldc + gc0;
    const size_t i1 = (size_t)grow * ldc + gc1;
    C[i0] = v0;
    C[i1] = v1;
    if (EP == EP_BIAS_ACC) {  // acc += X (each element owned by one lane)
      acc[i0] += v0;
      acc[i1] += v1;
    }
  }
}

// ---------------------------------------------------------------------------
// bf16 Gram kernel: S = Hb @ Hb^T (f32 accumulate) + fused global sum.
// Hb is [Nr x D] bf16 (2 MB -> fully L2 resident), so A/B fragments are
// loaded straight from global memory as 16B-aligned b128s following the
// 16-bit A/B WMMA VGPR layouts; no LDS staging needed.
// Wave computes a 16x32 strip (two 16x16 tiles); block tile 64x64.
// ---------------------------------------------------------------------------
__global__ __launch_bounds__(256) void gram_bf16(
    const unsigned short* __restrict__ Hb, float* __restrict__ S,
    float* __restrict__ gsum, int Nr, int D) {
  __shared__ float red[8];
  const int tid  = threadIdx.x;
  const int lane = tid & 31;
  const int wave = tid >> 5;
  const int wm   = wave & 3;
  const int wn   = wave >> 2;
  const int bm   = blockIdx.y * 64;
  const int bn   = blockIdx.x * 64;

  // A fragment (16-bit 16x32): lanes 0-15 -> K[0..7],[16..23]; 16-31 -> +8
  const int arow  = bm + wm * 16 + (lane & 15);
  const int abase = (lane >> 4) * 8;
  // B fragment (16-bit 32x16): lanes 0-15 -> K[0..7],[16..23]; 16-31 -> +8
  const int bcol0  = bn + wn * 32 + (lane & 15);
  const int bcol1  = bcol0 + 16;
  const int bkbase = (lane >> 4) * 8;

  union U { v16bf v; uint4 q[2]; };
  v8f c0 = {}, c1 = {};

  for (int k0 = 0; k0 < D; k0 += 32) {
    U a, b0, b1;
    const unsigned short* ap  = Hb + (size_t)arow * D + k0 + abase;
    a.q[0] = *(const uint4*)(ap);        // K  abase..abase+7
    a.q[1] = *(const uint4*)(ap + 16);   // K  abase+16..abase+23
    const unsigned short* bp0 = Hb + (size_t)bcol0 * D + k0 + bkbase;
    b0.q[0] = *(const uint4*)(bp0);
    b0.q[1] = *(const uint4*)(bp0 + 16);
    const unsigned short* bp1 = Hb + (size_t)bcol1 * D + k0 + bkbase;
    b1.q[0] = *(const uint4*)(bp1);
    b1.q[1] = *(const uint4*)(bp1 + 16);
    c0 = __builtin_amdgcn_wmma_f32_16x16x32_bf16(false, a.v, false, b0.v, (short)0, c0, false, false);
    c1 = __builtin_amdgcn_wmma_f32_16x16x32_bf16(false, a.v, false, b1.v, (short)0, c1, false, false);
  }

  // store S tile + block-level sum for the global mean
  const int row0 = bm + wm * 16;
  float lsum = 0.f;
#pragma unroll
  for (int r = 0; r < 8; ++r) {
    const int grow = row0 + r + ((lane >> 4) << 3);
    const size_t i0 = (size_t)grow * Nr + bcol0;
    const size_t i1 = (size_t)grow * Nr + bcol1;
    S[i0] = c0[r];
    S[i1] = c1[r];
    lsum += c0[r] + c1[r];
  }
#pragma unroll
  for (int off = 16; off > 0; off >>= 1) lsum += __shfl_down(lsum, off);
  if (lane == 0) red[wave] = lsum;
  __syncthreads();
  if (tid == 0) {
    float s = 0.f;
#pragma unroll
    for (int i = 0; i < 8; ++i) s += red[i];
    atomicAdd(gsum, s);
  }
}

// Round-to-nearest-even fp32 -> bf16.
__global__ void f32_to_bf16(const float* __restrict__ a,
                            unsigned short* __restrict__ b, int n) {
  const int i = blockIdx.x * blockDim.x + threadIdx.x;
  if (i < n) {
    const unsigned int u = __float_as_uint(a[i]);
    const unsigned int r = u + 0x7FFFu + ((u >> 16) & 1u);
    b[i] = (unsigned short)(r >> 16);
  }
}

// Per-column batch-stats (training-mode BN): mu[j], biased var[j].
__global__ __launch_bounds__(256) void colstats(const float* __restrict__ H,
                                                float* __restrict__ mu,
                                                float* __restrict__ var,
                                                int N, int D) {
  const int j = blockIdx.x;
  float s = 0.f, q = 0.f;
  for (int i = threadIdx.x; i < N; i += 256) {
    const float v = H[(size_t)i * D + j];
    s += v;
    q += v * v;
  }
  __shared__ float ss[256], qq[256];
  ss[threadIdx.x] = s;
  qq[threadIdx.x] = q;
  __syncthreads();
  for (int off = 128; off > 0; off >>= 1) {
    if (threadIdx.x < off) {
      ss[threadIdx.x] += ss[threadIdx.x + off];
      qq[threadIdx.x] += qq[threadIdx.x + off];
    }
    __syncthreads();
  }
  if (threadIdx.x == 0) {
    const float m = ss[0] / (float)N;
    mu[j]  = m;
    var[j] = qq[0] / (float)N - m * m;
  }
}

__global__ void copy_f32(const float* __restrict__ a, float* __restrict__ b, int n) {
  const int i = blockIdx.x * blockDim.x + threadIdx.x;
  if (i < n) b[i] = a[i];
}

__global__ void zero_f32(float* __restrict__ a, int n) {
  const int i = blockIdx.x * blockDim.x + threadIdx.x;
  if (i < n) a[i] = 0.f;
}

// out += scale * sum((a-b)^2) over this block's slice (d_out zeroed beforehand)
__global__ __launch_bounds__(256) void mse_scaled(const float* __restrict__ a,
                                                  const float* __restrict__ b,
                                                  float* __restrict__ out,
                                                  float scale, int n) {
  const int i = blockIdx.x * blockDim.x + threadIdx.x;
  float d = 0.f;
  if (i < n) {
    const float t = a[i] - b[i];
    d = t * t;
  }
  __shared__ float ss[256];
  ss[threadIdx.x] = d;
  __syncthreads();
  for (int off = 128; off > 0; off >>= 1) {
    if (threadIdx.x < off) ss[threadIdx.x] += ss[threadIdx.x + off];
    __syncthreads();
  }
  if (threadIdx.x == 0) atomicAdd(out, ss[0] * scale);
}

// ---------------------------------------------------------------------------

struct Branch {
  const float* aW;  // [2,D,D]
  const float* ab;  // [2,D]
  const float* mW;  // [L,2,D,D]
  const float* mb;  // [L,2,D]
  const float* bn;  // [L,2,D] (gamma,beta)
};

extern "C" void kernel_launch(void* const* d_in, const int* in_sizes, int n_in,
                              void* d_out, int out_size, void* d_ws, size_t ws_size,
                              hipStream_t stream) {
  constexpr int N = 4096, D = 256, L = 3;
  (void)in_sizes; (void)n_in; (void)out_size; (void)ws_size;

  const float* Xin[4] = {(const float*)d_in[0], (const float*)d_in[1],
                         (const float*)d_in[2], (const float*)d_in[3]};
  Branch br[3];
  for (int b = 0; b < 3; ++b) {
    const int base = 4 + 5 * b;
    br[b].aW = (const float*)d_in[base + 0];
    br[b].ab = (const float*)d_in[base + 1];
    br[b].mW = (const float*)d_in[base + 2];
    br[b].mb = (const float*)d_in[base + 3];
    br[b].bn = (const float*)d_in[base + 4];
  }
  const int bidx[4] = {0, 1, 2, 1};  // QV->qv, Q->t, V->v, A->t

  // workspace layout (floats)
  float* ws  = (float*)d_ws;
  size_t off = 0;
  float* S   = ws + off; off += (size_t)N * N;  // 64 MB
  float* H1  = ws + off; off += (size_t)N * D;
  float* H2  = ws + off; off += (size_t)N * D;
  float* X0  = ws + off; off += (size_t)N * D;
  float* X1  = ws + off; off += (size_t)N * D;
  float* acc[4];
  for (int g = 0; g < 4; ++g) { acc[g] = ws + off; off += (size_t)N * D; }
  unsigned short* Hb = (unsigned short*)(ws + off); off += (size_t)N * D / 2;
  float* mu   = ws + off; off += D;
  float* var  = ws + off; off += D;
  float* gsum = ws + off; off += 16;

  float* out = (float*)d_out;
  zero_f32<<<1, 32, 0, stream>>>(out, 3);

  const dim3 blk(256);
  const dim3 gLin(D / BLK_N, N / BLK_M);   // [4096 x 256] outputs
  const dim3 gGram(N / 64, N / 64);        // [4096 x 4096] output
  const int ND = N * D;
  const float invNN = 1.f / ((float)N * (float)N);

  for (int g = 0; g < 4; ++g) {
    const Branch& P = br[bidx[g]];
    copy_f32<<<ND / 256, blk, 0, stream>>>(Xin[g], X0, ND);
    zero_f32<<<ND / 256, blk, 0, stream>>>(acc[g], ND);
    float* Xc = X0;
    float* Xn = X1;
    for (int l = 0; l < L; ++l) {
      // H1 = elu(Xc @ aW0 + ab0)
      wmma_gemm<AM_DIRECT, BM_DIRECT, EP_BIAS_ELU><<<gLin, blk, 0, stream>>>(
          Xc, P.aW, H1, P.ab, nullptr, nullptr, nullptr, nullptr,
          nullptr, 0.f, nullptr, N, D, D, D, D, D);
      // H2 = elu(H1 @ aW1 + ab1)
      wmma_gemm<AM_DIRECT, BM_DIRECT, EP_BIAS_ELU><<<gLin, blk, 0, stream>>>(
          H1, P.aW + (size_t)D * D, H2, P.ab + D, nullptr, nullptr, nullptr, nullptr,
          nullptr, 0.f, nullptr, N, D, D, D, D, D);
      // S = H2 @ H2^T in bf16 WMMA (S only feeds the "> mean" threshold)
      zero_f32<<<1, 32, 0, stream>>>(gsum, 1);
      f32_to_bf16<<<ND / 256, blk, 0, stream>>>(H2, Hb, ND);
      gram_bf16<<<gGram, blk, 0, stream>>>(Hb, S, gsum, N, D);
      // H1 = (thresh(S) + E*I) @ Xc  (adj_onehot materialized on the fly, fp32)
      wmma_gemm<AM_THRESH, BM_DIRECT, EP_PLAIN><<<gLin, blk, 0, stream>>>(
          S, Xc, H1, nullptr, nullptr, nullptr, nullptr, nullptr,
          gsum, invNN, nullptr, N, D, N, N, D, D);
      // H2 = H1 @ mW0 + mb0
      const float* W0 = P.mW + (size_t)(l * 2 + 0) * D * D;
      const float* W1 = P.mW + (size_t)(l * 2 + 1) * D * D;
      const float* b0 = P.mb + (size_t)(l * 2 + 0) * D;
      const float* b1 = P.mb + (size_t)(l * 2 + 1) * D;
      const float* ga = P.bn + (size_t)(l * 2 + 0) * D;
      const float* be = P.bn + (size_t)(l * 2 + 1) * D;
      wmma_gemm<AM_DIRECT, BM_DIRECT, EP_BIAS><<<gLin, blk, 0, stream>>>(
          H1, W0, H2, b0, nullptr, nullptr, nullptr, nullptr,
          nullptr, 0.f, nullptr, N, D, D, D, D, D);
      // batch stats of H2
      colstats<<<D, blk, 0, stream>>>(H2, mu, var, N, D);
      // Xn = relu(bn(H2)) @ mW1 + mb1 ; acc += Xn   (BN+ReLU fused in A-load)
      wmma_gemm<AM_BN, BM_DIRECT, EP_BIAS_ACC><<<gLin, blk, 0, stream>>>(
          H2, W1, Xn, b1, mu, var, ga, be,
          nullptr, 0.f, acc[g], N, D, D, D, D, D);
      float* t = Xc; Xc = Xn; Xn = t;
    }
  }

  // mse(acc_x/L, acc_A/L) = sum((acc_x-acc_A)^2) / (N*D*L^2)
  const float scale = 1.f / ((float)L * (float)L * (float)N * (float)D);
  mse_scaled<<<ND / 256, blk, 0, stream>>>(acc[0], acc[3], out + 0, scale, ND);
  mse_scaled<<<ND / 256, blk, 0, stream>>>(acc[1], acc[3], out + 1, scale, ND);
  mse_scaled<<<ND / 256, blk, 0, stream>>>(acc[2], acc[3], out + 2, scale, ND);
}
